// GumbelVectorQuantizer_26001732009984
// MI455X (gfx1250) — compile-verified, matrix operand order checked by reading, not run on hardware
//
#include <hip/hip_runtime.h>

// ---------------------------------------------------------------------------
// GumbelVectorQuantizer fused forward for MI455X (gfx1250, wave32).
//   K0: split W[512,640] fp32 -> bf16 hi/lo in WMMA-fragment-contiguous order.
//   K1: per 32-token strip (2 M-tiles/wave): async-load hidden strip to LDS,
//       bf16x3 WMMA GEMM (hi*hi + hi*lo + lo*hi ~ fp32), logits in LDS,
//       then softmax(dist) + argmax(logits+gumbel) + codebook gather.
// Roofline: HBM floor ~270MB => ~11.6us @23.3TB/s. bf16 K=32 WMMA keeps the
// 21.5 GFLOP GEMM near that floor; 2 M-tiles per wave halves B-fragment L2
// traffic (2.7GB -> 1.3GB) so L2 doesn't become the binding constraint.
// ---------------------------------------------------------------------------

typedef __bf16 v16bf __attribute__((ext_vector_type(16)));
typedef float  v8f   __attribute__((ext_vector_type(8)));
typedef float  v4f   __attribute__((ext_vector_type(4)));

constexpr int  TOK  = 8 * 4096;   // B*S tokens
constexpr int  DIMK = 512;        // reduction dim
constexpr int  GVN  = 640;        // G*V columns
constexpr int  NG   = 2;          // groups
constexpr int  NV   = 320;        // vars per group
constexpr int  ND   = 128;        // codevector dim per group
constexpr long DIST_OFF = (long)TOK * (NG * ND);   // 8,388,608 floats

// ---- Kernel 0: W[k][n] -> fragment-ordered bf16 hi/lo ----------------------
// Fragment layout per column n, K-chunk c, lane-half h:
//   16 contiguous bf16 = { K=k0..k0+7, K=k0+16..k0+23 },  k0 = c*32 + h*8.
// A lane's whole B fragment is then ONE aligned 32B load (2x b128, no moves).
__global__ __launch_bounds__(256)
void split_w_kernel(const float* __restrict__ W,
                    __bf16* __restrict__ wt_hi,
                    __bf16* __restrict__ wt_lo) {
  int o = blockIdx.x * 256 + threadIdx.x;        // fragment-ordered index
  int j = o & 15;
  int h = (o >> 4) & 1;
  int c = (o >> 5) & 15;
  int n = o >> 9;
  int k = c * 32 + h * 8 + ((j < 8) ? j : j + 8);
  float w  = W[k * GVN + n];
  __bf16 hi = (__bf16)w;
  __bf16 lo = (__bf16)(w - (float)hi);
  wt_hi[o] = hi;
  wt_lo[o] = lo;
}

// ---- Kernel 1: fused GEMM + gumbel-VQ epilogue -----------------------------
__global__ __launch_bounds__(256)
void gvq_fused_kernel(const float* __restrict__ hs,
                      const __bf16* __restrict__ wt_hi,
                      const __bf16* __restrict__ wt_lo,
                      const float* __restrict__ bias,
                      const float* __restrict__ cb,     // [G*V, 128]
                      const float* __restrict__ gum,    // [T*G, V]
                      float* __restrict__ out) {        // [out | dist]
  // Phase A: smem = hidden strip [32][516] fp32 (padded rows: 16 GEMM lanes
  // hit 16 distinct bank offsets).  Phase B reuses it as logits [32][648].
  extern __shared__ float smem[];                  // 82,944 B (dynamic)

  const int tid  = threadIdx.x;
  const int lane = tid & 31;
  const int wave = tid >> 5;
  const int hf   = lane >> 4;          // lane half selects K sub-block
  const int l16  = lane & 15;
  const long tok0 = (long)blockIdx.x * 32;

  // ---- stage 32x512 fp32 hidden strip straight into LDS (ASYNCcnt path) ---
  {
    const float* src = hs + tok0 * DIMK;
    #pragma unroll
    for (int i = 0; i < 16; ++i) {
      int lin = (i * 256 + tid) * 4;               // 4096 float4 total
      int r = lin >> 9, k = lin & 511;
      unsigned dsaddr = (unsigned)((r * 516 + k) * 4);
      unsigned long long gaddr = (unsigned long long)(const void*)(src + lin);
      asm volatile("global_load_async_to_lds_b128 %0, %1, off"
                   :: "v"(dsaddr), "v"(gaddr) : "memory");
    }
    asm volatile("s_wait_asynccnt 0" ::: "memory");
  }
  __syncthreads();

  // ---- GEMM: wave owns 5 N-tiles x 2 M-tiles (B fragments feed 6 WMMAs) ---
  v8f acc[2][5];
  #pragma unroll
  for (int m = 0; m < 2; ++m)
    #pragma unroll
    for (int t = 0; t < 5; ++t)
      #pragma unroll
      for (int i = 0; i < 8; ++i) acc[m][t][i] = 0.0f;

  const int ntbase = wave * 5;

  for (int chunk = 0; chunk < 16; ++chunk) {
    const int k0 = chunk * 32 + hf * 8;            // ISA 16-bit 16x32 layout
    // A fragments for both M-tiles, split fp32 -> bf16 hi + lo on the fly.
    v16bf ahi[2], alo[2];
    #pragma unroll
    for (int m = 0; m < 2; ++m) {
      const float* ap = &smem[(m * 16 + l16) * 516 + k0];
      v4f a0 = *(const v4f*)(ap);      v4f a1 = *(const v4f*)(ap + 4);
      v4f a2 = *(const v4f*)(ap + 16); v4f a3 = *(const v4f*)(ap + 20);
      #pragma unroll
      for (int i = 0; i < 4; ++i) {
        float x;
        x = a0[i]; ahi[m][i]      = (__bf16)x; alo[m][i]      = (__bf16)(x - (float)ahi[m][i]);
        x = a1[i]; ahi[m][4 + i]  = (__bf16)x; alo[m][4 + i]  = (__bf16)(x - (float)ahi[m][4 + i]);
        x = a2[i]; ahi[m][8 + i]  = (__bf16)x; alo[m][8 + i]  = (__bf16)(x - (float)ahi[m][8 + i]);
        x = a3[i]; ahi[m][12 + i] = (__bf16)x; alo[m][12 + i] = (__bf16)(x - (float)ahi[m][12 + i]);
      }
    }
    #pragma unroll
    for (int t = 0; t < 5; ++t) {
      // B fragment: fragment-contiguous => one aligned 32B load each.
      const int n = (ntbase + t) * 16 + l16;
      const size_t foff = ((size_t)n * 32 + chunk * 2 + hf) * 16;
      v16bf bhi = *(const v16bf*)(wt_hi + foff);
      v16bf blo = *(const v16bf*)(wt_lo + foff);
      #pragma unroll
      for (int m = 0; m < 2; ++m) {
        acc[m][t] = __builtin_amdgcn_wmma_f32_16x16x32_bf16(false, ahi[m], false, bhi,
                                                            (short)0, acc[m][t], false, false);
        acc[m][t] = __builtin_amdgcn_wmma_f32_16x16x32_bf16(false, ahi[m], false, blo,
                                                            (short)0, acc[m][t], false, false);
        acc[m][t] = __builtin_amdgcn_wmma_f32_16x16x32_bf16(false, alo[m], false, bhi,
                                                            (short)0, acc[m][t], false, false);
      }
    }
  }

  __syncthreads();   // strip reads done; smem becomes logits [32][648]

  // C/D layout: lanes 0-15: N=lane, vgpr r -> M=r; lanes 16-31: M=r+8.
  #pragma unroll
  for (int t = 0; t < 5; ++t) {
    const int col = (ntbase + t) * 16 + l16;
    const float bb = bias[col];
    #pragma unroll
    for (int m = 0; m < 2; ++m)
      #pragma unroll
      for (int r = 0; r < 8; ++r) {
        const int row = m * 16 + r + hf * 8;
        smem[row * 648 + col] = acc[m][t][r] + bb;
      }
  }
  __syncthreads();

  // ---- epilogue: 64 (token,group) rows of V=320; 8 rows per wave ----------
  for (int rr = 0; rr < 8; ++rr) {
    const int row = wave * 8 + rr;
    const int tok = row >> 1, g = row & 1;
    const long tglob = tok0 + tok;
    const float* lg = &smem[tok * 648 + g * NV];

    float lv[10];
    float mx = -3.402823466e38f;
    #pragma unroll
    for (int j = 0; j < 10; ++j) { lv[j] = lg[lane + 32 * j]; mx = fmaxf(mx, lv[j]); }
    #pragma unroll
    for (int off = 16; off; off >>= 1) mx = fmaxf(mx, __shfl_xor(mx, off, 32));

    float ev[10], sum = 0.0f;
    #pragma unroll
    for (int j = 0; j < 10; ++j) { ev[j] = __expf(lv[j] - mx); sum += ev[j]; }
    #pragma unroll
    for (int off = 16; off; off >>= 1) sum += __shfl_xor(sum, off, 32);
    const float inv = 1.0f / sum;

    float* dst = out + DIST_OFF + (tglob * NG + g) * (long)NV;
    #pragma unroll
    for (int j = 0; j < 10; ++j) dst[lane + 32 * j] = ev[j] * inv;

    // argmax(logits + gumbel); straight-through => out = codevectors[argmax]
    const float* gr = gum + (tglob * NG + g) * (long)NV;
    float bv = -3.402823466e38f; int bi = NV;
    #pragma unroll
    for (int j = 0; j < 10; ++j) {
      const int v = lane + 32 * j;
      const float x = lv[j] + gr[v];
      if (x > bv || (x == bv && v < bi)) { bv = x; bi = v; }
    }
    #pragma unroll
    for (int off = 16; off; off >>= 1) {
      const float ov = __shfl_xor(bv, off, 32);
      const int   oi = __shfl_xor(bi, off, 32);
      if (ov > bv || (ov == bv && oi < bi)) { bv = ov; bi = oi; }
    }

    // gather 128-float codevector row: 32 lanes x one float4 each
    const float* cv = cb + ((long)g * NV + bi) * (long)ND;
    float* od = out + tglob * (long)(NG * ND) + g * ND;
    *(v4f*)(od + lane * 4) = *(const v4f*)(cv + lane * 4);
  }
}

// ---------------------------------------------------------------------------
extern "C" void kernel_launch(void* const* d_in, const int* in_sizes, int n_in,
                              void* d_out, int out_size, void* d_ws, size_t ws_size,
                              hipStream_t stream) {
  const float* hs  = (const float*)d_in[0];   // hidden_states [8,4096,512]
  const float* W   = (const float*)d_in[1];   // [512, 640]
  const float* b   = (const float*)d_in[2];   // [640]
  const float* cb  = (const float*)d_in[3];   // [640, 128]
  const float* gum = (const float*)d_in[4];   // [65536, 320]

  __bf16* wt_hi = (__bf16*)d_ws;                       // fragment-ordered
  __bf16* wt_lo = wt_hi + (size_t)GVN * DIMK;

  split_w_kernel<<<(GVN * DIMK) / 256, 256, 0, stream>>>(W, wt_hi, wt_lo);

  const size_t shbytes = 32 * 648 * sizeof(float);     // 82,944 B
  gvq_fused_kernel<<<TOK / 32, 256, shbytes, stream>>>(hs, wt_hi, wt_lo, b, cb,
                                                       gum, (float*)d_out);
}
